// SecondaryCapsule_19928648253907
// MI455X (gfx1250) — matrix operand order repaired and checked by reading, not compiled, not run on hardware
//
#include <hip/hip_runtime.h>

typedef __attribute__((ext_vector_type(16))) __bf16        v16bf;
typedef __attribute__((ext_vector_type(8)))  __bf16        v8bf;
typedef __attribute__((ext_vector_type(8)))  float         v8f;
typedef __attribute__((ext_vector_type(4)))  unsigned int  v4u;
typedef __attribute__((ext_vector_type(4)))  int           v4i;
typedef __attribute__((ext_vector_type(8)))  int           v8i;

#define B_   64
#define I_   2048
#define E_   32
#define O_   32
#define D_   32
#define OD_  1024
#define EPS_ 1e-7f

#define WROW_ 36   // 32 f32 per od row + 4 DWORD TDM pad -> conflict-free banks

// raw LDS byte offset of a shared-memory pointer (generic -> addrspace(3))
__device__ __forceinline__ unsigned lds_off(const void* p) {
  return (unsigned)(unsigned long long)(const __attribute__((address_space(3))) char*)p;
}

// Issue one TDM descriptor: contiguous 1024x32 f32 W[i] slice -> LDS,
// pad_enable: +4 DWORDs after every 32 DWORDs (row pitch 36 f32).
__device__ __forceinline__ void tdm_load_w(const float* gsrc, unsigned lds_addr) {
  const unsigned long long ga = (unsigned long long)gsrc;
  v4u g0;
  g0[0] = 1u;                                   // count=1, user descriptor
  g0[1] = lds_addr;                             // lds_addr
  g0[2] = (unsigned)ga;                         // global_addr[31:0]
  g0[3] = (unsigned)(ga >> 32) | (2u << 30);    // global_addr[56:32] | type=2
  v8i g1;
  g1[0] = (2 << 16)      // data_size = 4B
        | (1 << 20)      // pad_enable
        | (4 << 22)      // pad_interval: 32 DWORDs
        | (3 << 25);     // pad_amount: 4 DWORDs
  g1[1] = (int)(0x8000u << 16);                 // tensor_dim0 = 32768 (lo16 in hi half)
  g1[2] = (1 << 16);                            // tensor_dim0 hi=0 | tensor_dim1 = 1
  g1[3] = (int)(0x8000u << 16);                 // tensor_dim1 hi=0 | tile_dim0 = 32768
  g1[4] = 0;                                    // tile_dim1 = 0 (unused), tile_dim2 = 0
  g1[5] = 32768;                                // tensor_dim0_stride[31:0]
  g1[6] = 0;
  g1[7] = 0;
  const v4i gz = {};
#if __clang_major__ >= 23
  const v8i gz8 = {};
  __builtin_amdgcn_tensor_load_to_lds(g0, g1, gz, gz, gz8, 0);
#else
  __builtin_amdgcn_tensor_load_to_lds(g0, g1, gz, gz, 0);
#endif
}

// ---------------------------------------------------------------------------
// Kernel 1: S0[b,od] = sum_i pred[b,i,od],  pred = X[b,i,:] . W[i,od,:]
// 512 thr = 16 waves; wave w owns od-tiles [4w,4w+4) x all 4 b-tiles.
// W[i] slices streamed by the Tensor Data Mover, double-buffered in LDS;
// WMMA C operand accumulates over both e (K=32) and i (chained accumulate).
// ---------------------------------------------------------------------------
__global__ __launch_bounds__(512) void k_pred_sum(const float* __restrict__ X,
                                                  const float* __restrict__ W,
                                                  float* __restrict__ S0) {
  __shared__ __align__(16) float  wbuf[2][OD_ * WROW_];  // 2 x 147456 B
  __shared__ __align__(16) __bf16 xb[64][40];            // X[.,i,.] bf16, padded
  const int t = threadIdx.x, lane = t & 31, wave = t >> 5;
  const int m0 = lane & 15, hi = lane >> 4;

  v8f acc[4][4] = {};                                    // [odt][mt]
  const int i0 = blockIdx.x * 32;

  if (wave == 0) tdm_load_w(W + (size_t)i0 * OD_ * E_, lds_off(&wbuf[0][0]));

  for (int ii = 0; ii < 32; ++ii) {
    const int i = i0 + ii;

    // ---- stage X[0:64, i, 0:32] -> LDS bf16 (512 thr x 4 elems) ----
    const int sb = t >> 3, se = (t & 7) * 4;
    const float4 xv = *(const float4*)(X + ((size_t)sb * I_ + i) * E_ + se);
    __syncthreads();   // prior iter done reading xb and wbuf[(ii+1)&1]
    if (wave == 0) {
      if (ii + 1 < 32) {
        tdm_load_w(W + (size_t)(i + 1) * OD_ * E_, lds_off(&wbuf[(ii + 1) & 1][0]));
        __builtin_amdgcn_s_wait_tensorcnt(1);   // wbuf[ii&1] ready (prefetch in flight)
      } else {
        __builtin_amdgcn_s_wait_tensorcnt(0);   // drain last DMA
      }
    }
    xb[sb][se + 0] = (__bf16)xv.x; xb[sb][se + 1] = (__bf16)xv.y;
    xb[sb][se + 2] = (__bf16)xv.z; xb[sb][se + 3] = (__bf16)xv.w;
    __syncthreads();   // xb written, TDM buffer complete

    // ---- A fragments (16x32 bf16, contiguous k = 16h+s, same order as B) ----
    v16bf afr[4];
#pragma unroll
    for (int mt = 0; mt < 4; ++mt) {
      const __bf16* row = &xb[mt * 16 + m0][0];
      const v8bf lo  = *(const v8bf*)(row + hi * 16);       // K = 16h..16h+7
      const v8bf hi8 = *(const v8bf*)(row + hi * 16 + 8);   // K = 16h+8..16h+15
#pragma unroll
      for (int j = 0; j < 8; ++j) { afr[mt][j] = lo[j]; afr[mt][j + 8] = hi8[j]; }
    }

    // ---- B fragments from the TDM-staged LDS buffer + WMMA ----
    const float* wb = &wbuf[ii & 1][0];
#pragma unroll
    for (int odt = 0; odt < 4; ++odt) {
      const int od = (wave * 4 + odt) * 16 + m0;
      const float* wrow = wb + od * WROW_ + hi * 16;        // 16 consecutive e
      float wv[16];
      *(float4*)(wv + 0)  = *(const float4*)(wrow + 0);
      *(float4*)(wv + 4)  = *(const float4*)(wrow + 4);
      *(float4*)(wv + 8)  = *(const float4*)(wrow + 8);
      *(float4*)(wv + 12) = *(const float4*)(wrow + 12);
      v16bf bfr;
#pragma unroll
      for (int j = 0; j < 16; ++j) bfr[j] = (__bf16)wv[j];
#pragma unroll
      for (int mt = 0; mt < 4; ++mt)
        acc[odt][mt] = __builtin_amdgcn_wmma_f32_16x16x32_bf16(
            false, afr[mt], false, bfr, (short)0, acc[odt][mt], false, false);
    }
  }

  // ---- reduce across i-chunks via global f32 atomics ----
#pragma unroll
  for (int odt = 0; odt < 4; ++odt) {
    const int od = (wave * 4 + odt) * 16 + m0;
#pragma unroll
    for (int mt = 0; mt < 4; ++mt)
#pragma unroll
      for (int r = 0; r < 8; ++r) {
        const int b = mt * 16 + hi * 8 + r;      // C/D: m = r + 8*(lane/16)
        atomicAdd(&S0[(size_t)b * OD_ + od], acc[odt][mt][r]);
      }
  }
}

// ---------------------------------------------------------------------------
// Squash: V[bo,:] = (|s|^2/(1+|s|^2)) * s/sqrt(|s|^2+eps), s = S[bo,:]*scale
// ---------------------------------------------------------------------------
__global__ __launch_bounds__(256) void k_squash(const float* __restrict__ S,
                                                float* __restrict__ V, float scale) {
  const int lane = threadIdx.x & 31, wave = threadIdx.x >> 5;
  const int bo = blockIdx.x * 8 + wave;
  const float s = S[(size_t)bo * 32 + lane] * scale;
  float sq = s * s;
#pragma unroll
  for (int off = 16; off; off >>= 1) sq += __shfl_xor(sq, off, 32);
  const float coef = (sq / (1.f + sq)) * rsqrtf(sq + EPS_);
  V[(size_t)bo * 32 + lane] = coef * s;
}

// ---------------------------------------------------------------------------
// Kernel 3: recompute pred tiles, agreement a=pred.v1, softmax over o,
//           S2[b,od] += c[b,o]*pred[b,i,od]. pred lives only in LDS.
// ---------------------------------------------------------------------------
__global__ __launch_bounds__(512) void k_route(const float* __restrict__ X,
                                               const float* __restrict__ W,
                                               const float* __restrict__ V1,
                                               float* __restrict__ S2) {
  __shared__ __align__(16) __bf16 xb[64][40];
  __shared__ __align__(16) __bf16 predl[1024][72];   // [od][b], padded
  __shared__ __align__(16) __bf16 v1l[2048][32];     // [b*32+o][d]
  __shared__ __align__(16) float  cb[32][64];        // agreement -> softmax c
  const int t = threadIdx.x, lane = t & 31, wave = t >> 5;
  const int m0 = lane & 15, hi = lane >> 4;

  // cache v1 (256 KB fp32 -> 128 KB bf16) block-wide, once
  for (int k = t; k < (2048 * 32) / 4; k += 512) {
    const float4 v = *(const float4*)(V1 + (size_t)k * 4);
    __bf16* p = &v1l[0][0] + (size_t)k * 4;
    p[0] = (__bf16)v.x; p[1] = (__bf16)v.y; p[2] = (__bf16)v.z; p[3] = (__bf16)v.w;
  }

  v8f acc[4][4] = {};
  const int i0 = blockIdx.x * 32;

  for (int ii = 0; ii < 32; ++ii) {
    const int i = i0 + ii;

    const int sb = t >> 3, se = (t & 7) * 4;
    const float4 xv = *(const float4*)(X + ((size_t)sb * I_ + i) * E_ + se);
    __syncthreads();                              // also fences v1l on ii==0
    xb[sb][se + 0] = (__bf16)xv.x; xb[sb][se + 1] = (__bf16)xv.y;
    xb[sb][se + 2] = (__bf16)xv.z; xb[sb][se + 3] = (__bf16)xv.w;
    __syncthreads();

    v16bf afr[4];
#pragma unroll
    for (int mt = 0; mt < 4; ++mt) {
      const __bf16* row = &xb[mt * 16 + m0][0];
      const v8bf lo  = *(const v8bf*)(row + hi * 16);
      const v8bf hi8 = *(const v8bf*)(row + hi * 16 + 8);
#pragma unroll
      for (int j = 0; j < 8; ++j) { afr[mt][j] = lo[j]; afr[mt][j + 8] = hi8[j]; }
    }

    // ---- pred tiles via WMMA, spill to LDS (vector ds stores) ----
#pragma unroll
    for (int odt = 0; odt < 4; ++odt) {
      const int od = (wave * 4 + odt) * 16 + m0;
      const float* wp = W + ((size_t)i * OD_ + od) * E_ + hi * 16;
      float wv[16];
      *(float4*)(wv + 0)  = *(const float4*)(wp + 0);
      *(float4*)(wv + 4)  = *(const float4*)(wp + 4);
      *(float4*)(wv + 8)  = *(const float4*)(wp + 8);
      *(float4*)(wv + 12) = *(const float4*)(wp + 12);
      if (ii + 1 < 32) __builtin_prefetch(wp + (size_t)OD_ * E_, 0, 3);
      v16bf bfr;
#pragma unroll
      for (int j = 0; j < 16; ++j) bfr[j] = (__bf16)wv[j];
#pragma unroll
      for (int mt = 0; mt < 4; ++mt) {
        const v8f z = {};
        const v8f pt = __builtin_amdgcn_wmma_f32_16x16x32_bf16(
            false, afr[mt], false, bfr, (short)0, z, false, false);
        v8bf pb;
#pragma unroll
        for (int r = 0; r < 8; ++r) pb[r] = (__bf16)pt[r];
        *(v8bf*)&predl[od][mt * 16 + hi * 8] = pb;   // 8 consecutive b, 16B
      }
    }
    __syncthreads();

    // ---- agreement a[b,o] = sum_d pred[b,o*32+d] * v1[b,o,d] ----
#pragma unroll
    for (int p = 0; p < 4; ++p) {
      const int bo = t + p * 512;                 // covers 2048 pairs
      const int b = bo >> 5, o = bo & 31;
      const v8bf* vr = (const v8bf*)&v1l[bo][0];
      float s = 0.f;
#pragma unroll
      for (int dq = 0; dq < 4; ++dq) {
        const v8bf vv = vr[dq];
#pragma unroll
        for (int j = 0; j < 8; ++j)
          s += (float)predl[o * 32 + dq * 8 + j][b] * (float)vv[j];
      }
      cb[o][b] = s;
    }
    __syncthreads();

    // ---- softmax over o, per b (b_logits start at 0, ROUTINGS=2) ----
    if (t < 64) {
      float mx = -3.4e38f;
#pragma unroll
      for (int o = 0; o < 32; ++o) mx = fmaxf(mx, cb[o][t]);
      float ex[32], sum = 0.f;
#pragma unroll
      for (int o = 0; o < 32; ++o) { ex[o] = __expf(cb[o][t] - mx); sum += ex[o]; }
      const float inv = 1.f / sum;
#pragma unroll
      for (int o = 0; o < 32; ++o) cb[o][t] = ex[o] * inv;
    }
    __syncthreads();

    // ---- acc[b,od] += c[b,o] * pred[b,od]  (o fixed per od-tile) ----
#pragma unroll
    for (int odt = 0; odt < 4; ++odt) {
      const int od = (wave * 4 + odt) * 16 + m0;
      const int o = od >> 5;
#pragma unroll
      for (int mt = 0; mt < 4; ++mt) {
        const int b0 = mt * 16 + hi * 8;
        const float4 c0 = *(const float4*)&cb[o][b0];
        const float4 c1 = *(const float4*)&cb[o][b0 + 4];
        const v8bf pb = *(const v8bf*)&predl[od][b0];
        acc[odt][mt][0] += c0.x * (float)pb[0];
        acc[odt][mt][1] += c0.y * (float)pb[1];
        acc[odt][mt][2] += c0.z * (float)pb[2];
        acc[odt][mt][3] += c0.w * (float)pb[3];
        acc[odt][mt][4] += c1.x * (float)pb[4];
        acc[odt][mt][5] += c1.y * (float)pb[5];
        acc[odt][mt][6] += c1.z * (float)pb[6];
        acc[odt][mt][7] += c1.w * (float)pb[7];
      }
    }
  }

#pragma unroll
  for (int odt = 0; odt < 4; ++odt) {
    const int od = (wave * 4 + odt) * 16 + m0;
#pragma unroll
    for (int mt = 0; mt < 4; ++mt)
#pragma unroll
      for (int r = 0; r < 8; ++r) {
        const int b = mt * 16 + hi * 8 + r;
        atomicAdd(&S2[(size_t)b * OD_ + od], acc[odt][mt][r]);
      }
  }
}

// ---------------------------------------------------------------------------
extern "C" void kernel_launch(void* const* d_in, const int* in_sizes, int n_in,
                              void* d_out, int out_size, void* d_ws, size_t ws_size,
                              hipStream_t stream) {
  const float* X = (const float*)d_in[0];   // [64, 2048, 32]
  const float* W = (const float*)d_in[1];   // [2048, 32, 32, 32]
  float* S0 = (float*)d_ws;                 // [64,1024]
  float* S2 = S0 + (size_t)B_ * OD_;        // [64,1024]
  float* V1 = S2 + (size_t)B_ * OD_;        // [64,1024]

  (void)hipMemsetAsync(d_ws, 0, (size_t)2 * B_ * OD_ * sizeof(float), stream);

  k_pred_sum<<<I_ / 32, 512, 0, stream>>>(X, W, S0);
  k_squash  <<<(B_ * O_) / 8, 256, 0, stream>>>(S0, V1, 1.0f / 32.0f);
  k_route   <<<I_ / 32, 512, 0, stream>>>(X, W, V1, S2);
  k_squash  <<<(B_ * O_) / 8, 256, 0, stream>>>(S2, (float*)d_out, 1.0f);
}
